// Block_49744311222516
// MI455X (gfx1250) — matrix operand order repaired and checked by reading, not compiled, not run on hardware
//
#include <hip/hip_runtime.h>
#include <hip/hip_bf16.h>

// ---------------------------------------------------------------------------
// RWKV-4 block for MI455X (gfx1250, wave32, WMMA).
// GEMMs: C[M,N] = A[M,K] * W[N,K]^T, A/W bf16 (K-contiguous), f32 accum via
// v_wmma_f32_16x16x32_bf16. Staging uses gfx1250 async global->LDS DMA
// (global_load_async_to_lds_b128 + s_wait_asynccnt) with double buffering.
// ---------------------------------------------------------------------------

typedef __attribute__((ext_vector_type(16))) __bf16 v16bf;
typedef __attribute__((ext_vector_type(8)))  __bf16 bf16x8;
typedef __attribute__((ext_vector_type(4)))  __bf16 bf16x4;
typedef __attribute__((ext_vector_type(8)))  float  v8f;

union BF16Frag { v16bf v; bf16x8 h[2]; };

constexpr int BB = 8, TT = 2048, CC = 1024, FF = 4096;
constexpr int MM = BB * TT;                 // 16384 rows

// ------------------------------ GEMM -----------------------------------
constexpr int BMt = 128, BNt = 128, BKt = 32;
constexpr int LDT = BKt + 16;               // bf16 elems; 96B row stride (16B aligned)

__device__ __forceinline__ unsigned lds_addr32(const void* p) {
    // addrspace(3) generic pointers: LDS byte address = addr[31:0]
    return (unsigned)(unsigned long long)p;
}

__device__ __forceinline__ void async_ld_b128(unsigned lds, unsigned long long gaddr) {
    asm volatile("global_load_async_to_lds_b128 %0, %1, off"
                 :: "v"(lds), "v"(gaddr) : "memory");
}

__device__ __forceinline__ void wait_async0() {
    asm volatile("s_wait_asynccnt 0x0" ::: "memory");
}

// MODE 0: out f32 = acc
// MODE 1: out f32 = sigmoid(acc)
// MODE 2: out f32 = Res + acc
// MODE 3: out bf16 = relu(acc)^2
// MODE 4: out f32 = Res + Gate * acc
template <int MODE, int KD, int ND>
__global__ __launch_bounds__(256) void gemm_bf16_kernel(
    const __bf16* __restrict__ A, const __bf16* __restrict__ Bw,
    void* __restrict__ Outv,
    const float* __restrict__ Res, const float* __restrict__ Gate)
{
    __shared__ __align__(16) __bf16 As[2][BMt * LDT];
    __shared__ __align__(16) __bf16 Bs[2][BNt * LDT];

    const int tid  = threadIdx.x;
    const int lane = tid & 31;
    const int wid  = tid >> 5;            // 8 waves
    const int wm   = wid >> 2;            // 0..1 -> rows wm*64
    const int wn   = wid & 3;             // 0..3 -> cols wn*32
    const size_t mbase = (size_t)blockIdx.y * BMt;
    const int    nbase = blockIdx.x * BNt;

    v8f acc[4][2] = {};
    const int mrow  = lane & 15;
    const int ahalf = (lane >> 4) * 8;    // A frag: K quartering per ISA table
    const int bhalf = (lane >> 4) * 8;    // B frag: K quartering per ISA table

    // Per-thread staging chunks: 2 for A, 2 for B (16B each), 512 chunks/tile.
    const int r0  = (tid) >> 2;           // chunk row for c=0
    const int kc0 = ((tid) & 3) * 8;
    const int r1  = (tid + 256) >> 2;     // chunk row for c=1
    const int kc1 = ((tid + 256) & 3) * 8;

    auto issue_stage = [&](int buf, int k0) {
        async_ld_b128(lds_addr32(&As[buf][r0 * LDT + kc0]),
                      (unsigned long long)&A[(mbase + r0) * (size_t)KD + k0 + kc0]);
        async_ld_b128(lds_addr32(&Bs[buf][r0 * LDT + kc0]),
                      (unsigned long long)&Bw[((size_t)(nbase + r0)) * (size_t)KD + k0 + kc0]);
        async_ld_b128(lds_addr32(&As[buf][r1 * LDT + kc1]),
                      (unsigned long long)&A[(mbase + r1) * (size_t)KD + k0 + kc1]);
        async_ld_b128(lds_addr32(&Bs[buf][r1 * LDT + kc1]),
                      (unsigned long long)&Bw[((size_t)(nbase + r1)) * (size_t)KD + k0 + kc1]);
    };

    constexpr int NIT = KD / BKt;
    issue_stage(0, 0);

    for (int it = 0; it < NIT; ++it) {
        const int cur = it & 1;
        wait_async0();        // this wave's DMA into As/Bs[cur] landed in LDS
        __syncthreads();      // every wave's tile portion is visible
        if (it + 1 < NIT)     // prefetch next tile into the other buffer
            issue_stage(1 - cur, (it + 1) * BKt);

        v16bf afrag[4], bfrag[2];
#pragma unroll
        for (int mi = 0; mi < 4; mi++) {
            const int rr = wm * 64 + mi * 16 + mrow;
            BF16Frag u;
            u.h[0] = *(const bf16x8*)&As[cur][rr * LDT + ahalf];      // K 0..7 / 8..15
            u.h[1] = *(const bf16x8*)&As[cur][rr * LDT + 16 + ahalf]; // K 16..23 / 24..31
            afrag[mi] = u.v;
        }
#pragma unroll
        for (int ni = 0; ni < 2; ni++) {
            const int rr = wn * 32 + ni * 16 + mrow;
            BF16Frag u;
            u.h[0] = *(const bf16x8*)&Bs[cur][rr * LDT + bhalf];      // K 0..7 / 8..15
            u.h[1] = *(const bf16x8*)&Bs[cur][rr * LDT + bhalf + 16]; // K 16..23 / 24..31
            bfrag[ni] = u.v;
        }
#pragma unroll
        for (int mi = 0; mi < 4; mi++)
#pragma unroll
            for (int ni = 0; ni < 2; ni++)
                acc[mi][ni] = __builtin_amdgcn_wmma_f32_16x16x32_bf16(
                    false, afrag[mi], false, bfrag[ni],
                    (short)0, acc[mi][ni], false, false);
        __syncthreads();      // next iter's issue must not overwrite [cur] early
    }

    // Epilogue: C/D layout -> VGPR r holds row (r + 8*(lane>>4)), col = lane&15.
#pragma unroll
    for (int mi = 0; mi < 4; mi++) {
#pragma unroll
        for (int ni = 0; ni < 2; ni++) {
            const int    gcol  = nbase + wn * 32 + ni * 16 + (lane & 15);
            const size_t grow0 = mbase + wm * 64 + mi * 16 + (lane >> 4) * 8;
            const size_t base  = grow0 * (size_t)ND + gcol;
#pragma unroll
            for (int r = 0; r < 8; r++) {
                const size_t oidx = base + (size_t)r * ND;
                const float  va   = acc[mi][ni][r];
                if constexpr (MODE == 0) {
                    ((float*)Outv)[oidx] = va;
                } else if constexpr (MODE == 1) {
                    ((float*)Outv)[oidx] = 1.0f / (1.0f + __expf(-va));
                } else if constexpr (MODE == 2) {
                    ((float*)Outv)[oidx] = Res[oidx] + va;
                } else if constexpr (MODE == 3) {
                    const float rl = fmaxf(va, 0.0f);
                    ((__bf16*)Outv)[oidx] = (__bf16)(rl * rl);
                } else {
                    ((float*)Outv)[oidx] = Res[oidx] + Gate[oidx] * va;
                }
            }
        }
    }
}

// ------------------------ LayerNorm + shift + mix ------------------------
// One block per (b,t) row; recompute LN of row t-1 (no temp buffer needed).
__global__ __launch_bounds__(256) void prep_att_kernel(
    const float* __restrict__ x,
    const float* __restrict__ g,  const float* __restrict__ be,
    const float* __restrict__ mk, const float* __restrict__ mv,
    const float* __restrict__ mr,
    __bf16* __restrict__ oxk, __bf16* __restrict__ oxv, __bf16* __restrict__ oxr)
{
    const int row = blockIdx.x;
    const int t   = row & (TT - 1);
    const int tid = threadIdx.x;
    const int c0  = tid * 4;
    const size_t rb = (size_t)row * CC;

    float cur[4], prv[4];
    {
        float4 c4 = ((const float4*)(x + rb))[tid];
        cur[0] = c4.x; cur[1] = c4.y; cur[2] = c4.z; cur[3] = c4.w;
        if (t > 0) {
            float4 p4 = ((const float4*)(x + rb - CC))[tid];
            prv[0] = p4.x; prv[1] = p4.y; prv[2] = p4.z; prv[3] = p4.w;
        } else {
            prv[0] = prv[1] = prv[2] = prv[3] = 0.0f;
        }
    }
    __shared__ float4 red[256];
    float4 part;
    part.x = cur[0] + cur[1] + cur[2] + cur[3];
    part.y = cur[0]*cur[0] + cur[1]*cur[1] + cur[2]*cur[2] + cur[3]*cur[3];
    part.z = prv[0] + prv[1] + prv[2] + prv[3];
    part.w = prv[0]*prv[0] + prv[1]*prv[1] + prv[2]*prv[2] + prv[3]*prv[3];
    red[tid] = part;
    __syncthreads();
    for (int s = 128; s > 0; s >>= 1) {
        if (tid < s) {
            float4 o = red[tid + s];
            red[tid].x += o.x; red[tid].y += o.y;
            red[tid].z += o.z; red[tid].w += o.w;
        }
        __syncthreads();
    }
    const float4 tot = red[0];
    const float inv = 1.0f / CC;
    const float mc = tot.x * inv, vc = tot.y * inv - mc * mc;
    const float mp = tot.z * inv, vp = tot.w * inv - mp * mp;
    const float rc = rsqrtf(vc + 1e-5f), rp = rsqrtf(vp + 1e-5f);

    bf16x4 ok, ov, orr;
#pragma unroll
    for (int i = 0; i < 4; i++) {
        const int c = c0 + i;
        const float gg = g[c], bb = be[c];
        const float xn = (cur[i] - mc) * rc * gg + bb;
        const float xx = (t > 0) ? ((prv[i] - mp) * rp * gg + bb) : 0.0f;
        const float k_ = mk[c], v_ = mv[c], r_ = mr[c];
        ok[i]  = (__bf16)(xn * k_ + xx * (1.0f - k_));
        ov[i]  = (__bf16)(xn * v_ + xx * (1.0f - v_));
        orr[i] = (__bf16)(xn * r_ + xx * (1.0f - r_));
    }
    *(bf16x4*)(oxk + rb + c0) = ok;
    *(bf16x4*)(oxv + rb + c0) = ov;
    *(bf16x4*)(oxr + rb + c0) = orr;
}

__global__ __launch_bounds__(256) void prep_ffn_kernel(
    const float* __restrict__ x,
    const float* __restrict__ g,  const float* __restrict__ be,
    const float* __restrict__ mk, const float* __restrict__ mr,
    __bf16* __restrict__ oxk, __bf16* __restrict__ oxr)
{
    const int row = blockIdx.x;
    const int t   = row & (TT - 1);
    const int tid = threadIdx.x;
    const int c0  = tid * 4;
    const size_t rb = (size_t)row * CC;

    float cur[4], prv[4];
    {
        float4 c4 = ((const float4*)(x + rb))[tid];
        cur[0] = c4.x; cur[1] = c4.y; cur[2] = c4.z; cur[3] = c4.w;
        if (t > 0) {
            float4 p4 = ((const float4*)(x + rb - CC))[tid];
            prv[0] = p4.x; prv[1] = p4.y; prv[2] = p4.z; prv[3] = p4.w;
        } else {
            prv[0] = prv[1] = prv[2] = prv[3] = 0.0f;
        }
    }
    __shared__ float4 red[256];
    float4 part;
    part.x = cur[0] + cur[1] + cur[2] + cur[3];
    part.y = cur[0]*cur[0] + cur[1]*cur[1] + cur[2]*cur[2] + cur[3]*cur[3];
    part.z = prv[0] + prv[1] + prv[2] + prv[3];
    part.w = prv[0]*prv[0] + prv[1]*prv[1] + prv[2]*prv[2] + prv[3]*prv[3];
    red[tid] = part;
    __syncthreads();
    for (int s = 128; s > 0; s >>= 1) {
        if (tid < s) {
            float4 o = red[tid + s];
            red[tid].x += o.x; red[tid].y += o.y;
            red[tid].z += o.z; red[tid].w += o.w;
        }
        __syncthreads();
    }
    const float4 tot = red[0];
    const float inv = 1.0f / CC;
    const float mc = tot.x * inv, vc = tot.y * inv - mc * mc;
    const float mp = tot.z * inv, vp = tot.w * inv - mp * mp;
    const float rc = rsqrtf(vc + 1e-5f), rp = rsqrtf(vp + 1e-5f);

    bf16x4 ok, orr;
#pragma unroll
    for (int i = 0; i < 4; i++) {
        const int c = c0 + i;
        const float gg = g[c], bb = be[c];
        const float xn = (cur[i] - mc) * rc * gg + bb;
        const float xx = (t > 0) ? ((prv[i] - mp) * rp * gg + bb) : 0.0f;
        const float k_ = mk[c], r_ = mr[c];
        ok[i]  = (__bf16)(xn * k_ + xx * (1.0f - k_));
        orr[i] = (__bf16)(xn * r_ + xx * (1.0f - r_));
    }
    *(bf16x4*)(oxk + rb + c0) = ok;
    *(bf16x4*)(oxr + rb + c0) = orr;
}

// ------------------------------ WKV scan ---------------------------------
// One thread per (b,c) channel: 8192 channels = 256 wave32 waves.
__global__ __launch_bounds__(256) void wkv_kernel(
    const float* __restrict__ td, const float* __restrict__ tf,
    const float* __restrict__ K,  const float* __restrict__ V,
    const float* __restrict__ SR, __bf16* __restrict__ out)
{
    const int idx = blockIdx.x * 256 + threadIdx.x;   // < BB*CC
    const int b = idx / CC, c = idx % CC;
    const float w = -__expf(td[c]);
    const float u = tf[c];
    float aa = 0.0f, bb = 0.0f, pp = -1e38f;
    size_t base = (size_t)b * TT * CC + c;
    for (int t = 0; t < TT; t++, base += CC) {
        const float kt = K[base], vt = V[base], r = SR[base];
        const float ww = u + kt;
        const float p  = fmaxf(pp, ww);
        const float e1 = __expf(pp - p), e2 = __expf(ww - p);
        const float y  = (e1 * aa + e2 * vt) / (e1 * bb + e2);
        out[base] = (__bf16)(r * y);
        const float ww2 = pp + w;
        const float p2  = fmaxf(ww2, kt);
        const float e1b = __expf(ww2 - p2), e2b = __expf(kt - p2);
        aa = e1b * aa + e2b * vt;
        bb = e1b * bb + e2b;
        pp = p2;
    }
}

// --------------------------- weight convert ------------------------------
__global__ __launch_bounds__(256) void f32_to_bf16_kernel(
    const float* __restrict__ in, __bf16* __restrict__ out, int n)
{
    const int i = blockIdx.x * 256 + threadIdx.x;
    if (i < n) out[i] = (__bf16)in[i];
}

// ------------------------------- driver ----------------------------------
extern "C" void kernel_launch(void* const* d_in, const int* in_sizes, int n_in,
                              void* d_out, int out_size, void* d_ws, size_t ws_size,
                              hipStream_t stream) {
    const float* x    = (const float*)d_in[0];
    const float* ln1g = (const float*)d_in[1];
    const float* ln1b = (const float*)d_in[2];
    const float* ln2g = (const float*)d_in[3];
    const float* ln2b = (const float*)d_in[4];
    const float* amk  = (const float*)d_in[5];
    const float* amv  = (const float*)d_in[6];
    const float* amr  = (const float*)d_in[7];
    const float* td   = (const float*)d_in[8];
    const float* tf   = (const float*)d_in[9];
    const float* Wk   = (const float*)d_in[10];
    const float* Wv   = (const float*)d_in[11];
    const float* Wr   = (const float*)d_in[12];
    const float* Wo   = (const float*)d_in[13];
    const float* fmk  = (const float*)d_in[14];
    const float* fmr  = (const float*)d_in[15];
    const float* Wkf  = (const float*)d_in[16];  // [FFN, C]
    const float* Wrf  = (const float*)d_in[17];  // [C, C]
    const float* Wvf  = (const float*)d_in[18];  // [C, FFN]

    // Workspace arena (lifetime-overlapped; ~378 MiB total).
    char* ws = (char*)d_ws;
    const size_t MiB = 1ull << 20;
    __bf16* A0  = (__bf16*)(ws + 0);          // xk  -> rwkv        (32 MiB)
    __bf16* A1  = (__bf16*)(ws + 32 * MiB);   // xv  -> xk2         (32 MiB)
    __bf16* A2  = (__bf16*)(ws + 64 * MiB);   // xr  -> xr2         (32 MiB)
    float*  Kb  = (float*)(ws + 96 * MiB);    // k   -> x_res       (64 MiB)
    float*  Vb  = (float*)(ws + 160 * MiB);   // v   -> sigmoid(r2) (64 MiB)
    float*  SRb = (float*)(ws + 224 * MiB);   // sigmoid(r)         (64 MiB)
    __bf16* KF  = (__bf16*)(ws + 224 * MiB);  // kf (overlays SRb) (128 MiB)
    __bf16* wkB  = (__bf16*)(ws + 352 * MiB); // bf16 weights       (26 MiB)
    __bf16* wvB  = wkB  + (size_t)CC * CC;
    __bf16* wrB  = wvB  + (size_t)CC * CC;
    __bf16* woB  = wrB  + (size_t)CC * CC;
    __bf16* wkfB = woB  + (size_t)CC * CC;
    __bf16* wvfB = wkfB + (size_t)FF * CC;
    __bf16* wrfB = wvfB + (size_t)FF * CC;

    const int nCC = CC * CC, nFC = FF * CC;
    f32_to_bf16_kernel<<<(nCC + 255) / 256, 256, 0, stream>>>(Wk,  wkB,  nCC);
    f32_to_bf16_kernel<<<(nCC + 255) / 256, 256, 0, stream>>>(Wv,  wvB,  nCC);
    f32_to_bf16_kernel<<<(nCC + 255) / 256, 256, 0, stream>>>(Wr,  wrB,  nCC);
    f32_to_bf16_kernel<<<(nCC + 255) / 256, 256, 0, stream>>>(Wo,  woB,  nCC);
    f32_to_bf16_kernel<<<(nFC + 255) / 256, 256, 0, stream>>>(Wkf, wkfB, nFC);
    f32_to_bf16_kernel<<<(nFC + 255) / 256, 256, 0, stream>>>(Wvf, wvfB, nFC);
    f32_to_bf16_kernel<<<(nCC + 255) / 256, 256, 0, stream>>>(Wrf, wrfB, nCC);

    const dim3 blk(256);
    const dim3 gC(CC / BNt, MM / BMt);   // (8, 128)
    const dim3 gF(FF / BNt, MM / BMt);   // (32, 128)

    // --- TimeMix ---
    prep_att_kernel<<<MM, blk, 0, stream>>>(x, ln1g, ln1b, amk, amv, amr, A0, A1, A2);
    gemm_bf16_kernel<0, CC, CC><<<gC, blk, 0, stream>>>(A0, wkB, Kb,  nullptr, nullptr);
    gemm_bf16_kernel<0, CC, CC><<<gC, blk, 0, stream>>>(A1, wvB, Vb,  nullptr, nullptr);
    gemm_bf16_kernel<1, CC, CC><<<gC, blk, 0, stream>>>(A2, wrB, SRb, nullptr, nullptr);
    wkv_kernel<<<(BB * CC) / 256, blk, 0, stream>>>(td, tf, Kb, Vb, SRb, A0);
    gemm_bf16_kernel<2, CC, CC><<<gC, blk, 0, stream>>>(A0, woB, Kb, x, nullptr);

    // --- ChannelMix ---
    prep_ffn_kernel<<<MM, blk, 0, stream>>>(Kb, ln2g, ln2b, fmk, fmr, A1, A2);
    gemm_bf16_kernel<3, CC, FF><<<gF, blk, 0, stream>>>(A1, wkfB, KF, nullptr, nullptr);
    gemm_bf16_kernel<1, CC, CC><<<gC, blk, 0, stream>>>(A2, wrfB, Vb, nullptr, nullptr);
    gemm_bf16_kernel<4, FF, CC><<<gC, blk, 0, stream>>>(KF, wvfB, (float*)d_out, Kb, Vb);
}
